// Net_89541478187770
// MI455X (gfx1250) — compile-verified, matrix operand order checked by reading, not run on hardware
//
#include <hip/hip_runtime.h>
#include <hip/hip_bf16.h>

// ---------------------------------------------------------------------------
// CDNA5 (gfx1250) AttentiveFP implementation.
// All GEMMs use v_wmma_f32_16x16x32_bf16 (fp32->bf16 inputs, f32 accum).
// ---------------------------------------------------------------------------

typedef __attribute__((ext_vector_type(16))) __bf16    v16bf;
typedef __attribute__((ext_vector_type(8)))  __bf16    v8bf;
typedef __attribute__((ext_vector_type(8)))  float     v8f;
typedef __attribute__((ext_vector_type(8)))  unsigned  v8u;

#define NNODES 150000
#define NEDGES 300000
#define NGRAPH 4096
#define DHID   96
#define WSTRIDE 40   // LDS row stride in bf16 (padded 32->40: 80B, 16B-aligned, bank-rotating)

__device__ __forceinline__ unsigned bfbits(float f) {   // f32 -> bf16 bits (RNE)
    unsigned u = __builtin_bit_cast(unsigned, f);
    u += 0x7FFFu + ((u >> 16) & 1u);
    return u >> 16;
}
__device__ __forceinline__ __bf16 f2bf(float f) {
    unsigned short h = (unsigned short)bfbits(f);
    return __builtin_bit_cast(__bf16, h);
}

// monotone float <-> u32 encoding for atomicMax-based segment max
__device__ __forceinline__ unsigned fenc(float f) {
    unsigned u = __builtin_bit_cast(unsigned, f);
    return (u & 0x80000000u) ? ~u : (u | 0x80000000u);
}
__device__ __forceinline__ float fdec(unsigned u) {
    unsigned v = (u & 0x80000000u) ? (u & 0x7FFFFFFFu) : ~u;
    return __builtin_bit_cast(float, v);
}

__device__ __forceinline__ float sigmf(float x) { return 1.f / (1.f + expf(-x)); }

// ---------------------------------------------------------------------------
// Generic WMMA GEMM:  C[m, c0+n] = post( A'[m,:] @ W[n,:]^T + bias[n] )
//   A'[m,k] = pre( Araw(m,k) + abias[k] )
//   amode 0: Araw = A[m*lda + k]
//   amode 1: Araw = A[rowidx[m]*lda + k]                (edge gather)
//   amode 2: Araw = k<96 ? A[rowidx[m]*96+k] : EA[m*4+k-96]  (gather+concat)
// pre : 0 none, 1 elu      post: 0 none, 1 leaky(0.01), 2 relu, 3 elu
// Requires M % 16 == 0 (true for all call sites: 150000/300000/4096).
// Per-wave 16x64 output tile (4 wmma accums), 8 waves -> 128x64 block tile.
// W tile staged in LDS as bf16 [64 n][32 k], padded row stride (bank-rotating).
// Fully in-range K-steps use unconditional b128 loads; tails use branchless
// clamped loads + select (no exec manipulation).
// ---------------------------------------------------------------------------
__global__ void __launch_bounds__(256)
k_gemm(const float* __restrict__ A, const int* __restrict__ rowidx,
       const float* __restrict__ EA, const float* __restrict__ abias,
       const float* __restrict__ W, const float* __restrict__ bias,
       float* __restrict__ C,
       int M, int Nn, int K, int lda, int ldc, int c0,
       int amode, int preact, int postact)
{
    __shared__ __align__(16) __bf16 wlds[64 * WSTRIDE];

    const int tid  = threadIdx.x;
    const int wid  = tid >> 5;
    const int lane = tid & 31;
    const int n0   = blockIdx.x * 64;
    const int m0   = blockIdx.y * 128 + wid * 16;
    const bool active = (m0 < M);
    const int mrow = m0 + (lane & 15);
    const int kb   = (lane >> 4) * 8;   // wave32 A/B fragment half-select

    // hoisted A row pointer (gather index is loop-invariant)
    const float* ap = nullptr;
    if (active) {
        if (amode == 0) ap = A + (long)mrow * lda;
        else            ap = A + (long)rowidx[mrow] * lda;
    }

    // W staging coordinates for this thread
    const int sn   = tid >> 2;           // 0..63
    const int skc  = (tid & 3) * 8;      // 0,8,16,24
    const int sgn  = n0 + sn;
    const long sgc = (long)min(sgn, Nn - 1);   // clamped row (branchless OOB)

    v8f acc[4] = {};

    for (int kk = 0; kk < K; kk += 32) {
        const bool fullk = (kk + 32 <= K);
        // ---- cooperative W tile -> LDS (bf16) ----
        if (sgn < Nn && fullk) {
            const float* wp = W + sgc * K + kk + skc;
            float4 a = *(const float4*)wp;
            float4 b = *(const float4*)(wp + 4);
            uint4 pk;
            pk.x = bfbits(a.x) | (bfbits(a.y) << 16);
            pk.y = bfbits(a.z) | (bfbits(a.w) << 16);
            pk.z = bfbits(b.x) | (bfbits(b.y) << 16);
            pk.w = bfbits(b.z) | (bfbits(b.w) << 16);
            *(uint4*)&wlds[sn * WSTRIDE + skc] = pk;
            if (kk + 32 < K) __builtin_prefetch(wp + 32, 0, 3);  // next K-slice
        } else {
            // branchless clamped loads: always in-bounds, select 0 after
            #pragma unroll
            for (int j = 0; j < 8; ++j) {
                int gk = kk + skc + j;
                float w = W[sgc * K + min(gk, K - 1)];
                w = (sgn < Nn && gk < K) ? w : 0.f;
                wlds[sn * WSTRIDE + skc + j] = f2bf(w);
            }
        }
        __syncthreads();

        if (active) {
            // ---- gather 16 A elements (frag order: k = kb+i | 16+kb+i-8) ----
            float av[16];
            const bool fast = fullk && (amode != 2 || kk + 32 <= 96);
            if (fast) {
                float4 q0 = *(const float4*)(ap + kk + kb);
                float4 q1 = *(const float4*)(ap + kk + kb + 4);
                float4 q2 = *(const float4*)(ap + kk + 16 + kb);
                float4 q3 = *(const float4*)(ap + kk + 16 + kb + 4);
                av[0]=q0.x; av[1]=q0.y; av[2]=q0.z; av[3]=q0.w;
                av[4]=q1.x; av[5]=q1.y; av[6]=q1.z; av[7]=q1.w;
                av[8]=q2.x; av[9]=q2.y; av[10]=q2.z; av[11]=q2.w;
                av[12]=q3.x; av[13]=q3.y; av[14]=q3.z; av[15]=q3.w;
                if (abias) {
                    float4 b0 = *(const float4*)(abias + kk + kb);
                    float4 b1 = *(const float4*)(abias + kk + kb + 4);
                    float4 b2 = *(const float4*)(abias + kk + 16 + kb);
                    float4 b3 = *(const float4*)(abias + kk + 16 + kb + 4);
                    av[0]+=b0.x; av[1]+=b0.y; av[2]+=b0.z; av[3]+=b0.w;
                    av[4]+=b1.x; av[5]+=b1.y; av[6]+=b1.z; av[7]+=b1.w;
                    av[8]+=b2.x; av[9]+=b2.y; av[10]+=b2.z; av[11]+=b2.w;
                    av[12]+=b3.x; av[13]+=b3.y; av[14]+=b3.z; av[15]+=b3.w;
                }
            } else {
                // branchless tail: clamped index, pointer-select for concat region
                #pragma unroll
                for (int i = 0; i < 16; ++i) {
                    int k = kk + ((i < 8) ? (kb + i) : (8 + kb + i));
                    const float* bp = (amode == 2 && k >= 96)
                                          ? (EA + (long)mrow * 4 - 96) : ap;
                    int kc = min(k, K - 1);
                    float v = bp[kc];
                    if (abias) v += abias[kc];
                    av[i] = (k < K) ? v : 0.f;
                }
            }
            if (preact == 1) {
                #pragma unroll
                for (int i = 0; i < 16; ++i)
                    av[i] = (av[i] > 0.f) ? av[i] : (expf(av[i]) - 1.f);   // elu
            }
            // pack 16 floats -> bf16 fragment (pair-packed dwords)
            v8u apack;
            #pragma unroll
            for (int i = 0; i < 8; ++i)
                apack[i] = bfbits(av[2 * i]) | (bfbits(av[2 * i + 1]) << 16);
            v16bf afrag = __builtin_bit_cast(v16bf, apack);

            // ---- 4 B fragments from LDS, accumulate ----
            #pragma unroll
            for (int t = 0; t < 4; ++t) {
                int n = t * 16 + (lane & 15);
                v8bf lo = *(const v8bf*)&wlds[n * WSTRIDE + kb];
                v8bf hi = *(const v8bf*)&wlds[n * WSTRIDE + 16 + kb];
                v16bf bfrag = __builtin_shufflevector(lo, hi,
                                  0, 1, 2, 3, 4, 5, 6, 7, 8, 9, 10, 11, 12, 13, 14, 15);
                acc[t] = __builtin_amdgcn_wmma_f32_16x16x32_bf16(
                             false, afrag, false, bfrag, (short)0, acc[t], false, false);
            }
        }
        __syncthreads();
    }

    if (active) {
        int mbase = m0 + ((lane >> 4) << 3);
        #pragma unroll
        for (int t = 0; t < 4; ++t) {
            int n = n0 + t * 16 + (lane & 15);
            if (n < Nn) {
                float b = bias ? bias[n] : 0.f;
                #pragma unroll
                for (int r = 0; r < 8; ++r) {
                    float x = acc[t][r] + b;
                    if (postact == 1)      x = (x > 0.f) ? x : 0.01f * x;
                    else if (postact == 2) x = fmaxf(x, 0.f);
                    else if (postact == 3) x = (x > 0.f) ? x : (expf(x) - 1.f);
                    C[(long)(mbase + r) * ldc + c0 + n] = x;
                }
            }
        }
    }
}

// ---------------------------------------------------------------------------
// Elementwise / graph kernels
// ---------------------------------------------------------------------------
__global__ void k_fill_f(float* p, float v, long n) {
    long i = (long)blockIdx.x * blockDim.x + threadIdx.x;
    if (i < n) p[i] = v;
}
__global__ void k_fill_u(unsigned* p, unsigned v, long n) {
    long i = (long)blockIdx.x * blockDim.x + threadIdx.x;
    if (i < n) p[i] = v;
}
__global__ void k_relu(float* p, long n) {
    long i = (long)blockIdx.x * blockDim.x + threadIdx.x;
    if (i < n) p[i] = fmaxf(p[i], 0.f);
}

// wave-per-row: y1[r]=X[r,:]·v1, y2[r]=X[r,:]·v2 (v2 optional)
__global__ void k_rowdot2(const float* __restrict__ X, const float* __restrict__ v1,
                          const float* __restrict__ v2, float* y1, float* y2, int rows)
{
    int row  = blockIdx.x * (blockDim.x >> 5) + (threadIdx.x >> 5);
    int lane = threadIdx.x & 31;
    if (row >= rows) return;
    float s1 = 0.f, s2 = 0.f;
    for (int k = lane; k < DHID; k += 32) {
        float x = X[(long)row * DHID + k];
        s1 += x * v1[k];
        if (v2) s2 += x * v2[k];
    }
    #pragma unroll
    for (int off = 16; off > 0; off >>= 1) {
        s1 += __shfl_down(s1, off, 32);
        if (v2) s2 += __shfl_down(s2, off, 32);
    }
    if (lane == 0) { y1[row] = s1; if (y2) y2[row] = s2; }
}

// GATE conv score: alpha[e] = leaky(hj[e,:]·att_l + xr[dst[e]]), seg-max via atomicMax
__global__ void k_gate_score(const float* __restrict__ hj, const float* __restrict__ attl,
                             const float* __restrict__ xr, const int* __restrict__ dst,
                             float* alpha, unsigned* segmax, int E)
{
    int e    = blockIdx.x * (blockDim.x >> 5) + (threadIdx.x >> 5);
    int lane = threadIdx.x & 31;
    if (e >= E) return;
    float s = 0.f;
    for (int k = lane; k < DHID; k += 32) s += hj[(long)e * DHID + k] * attl[k];
    #pragma unroll
    for (int off = 16; off > 0; off >>= 1) s += __shfl_down(s, off, 32);
    if (lane == 0) {
        int d = dst[e];
        float a = s + xr[d];
        a = (a > 0.f) ? a : 0.01f * a;
        alpha[e] = a;
        atomicMax(&segmax[d], fenc(a));
    }
}

// GAT score: alpha[e] = leaky(ssrc[src[e]] + sdst[dst[e]])
__global__ void k_gat_score(const float* __restrict__ ssrc, const float* __restrict__ sdst,
                            const int* __restrict__ src, const int* __restrict__ dst,
                            float* alpha, unsigned* segmax, int E)
{
    int e = blockIdx.x * blockDim.x + threadIdx.x;
    if (e >= E) return;
    float a = ssrc[src[e]] + sdst[dst[e]];
    a = (a > 0.f) ? a : 0.01f * a;
    alpha[e] = a;
    atomicMax(&segmax[dst[e]], fenc(a));
}

// alpha[e] <- exp(alpha[e]-max[dst]); segsum[dst] += alpha[e]
__global__ void k_expsum(float* alpha, const unsigned* __restrict__ segmax,
                         const int* __restrict__ dst, float* segsum, int E)
{
    int e = blockIdx.x * blockDim.x + threadIdx.x;
    if (e >= E) return;
    int d = dst[e];
    float ex = expf(alpha[e] - fdec(segmax[d]));
    alpha[e] = ex;
    atomicAdd(&segsum[d], ex);
}

// out[dst[e],d] += msg * alpha[e]/(segsum[dst]+eps); msg from edge buf or xs[src]
__global__ void k_scatter(const float* __restrict__ medge, const float* __restrict__ mnode,
                          const int* __restrict__ src, const int* __restrict__ dst,
                          const float* __restrict__ alpha, const float* __restrict__ segsum,
                          float* out, long total)
{
    long i = (long)blockIdx.x * blockDim.x + threadIdx.x;
    if (i >= total) return;
    int e = (int)(i / DHID), d = (int)(i % DHID);
    float v = medge ? medge[i] : mnode[(long)src[e] * DHID + d];
    float w = alpha[e] / (segsum[dst[e]] + 1e-16f);
    atomicAdd(&out[(long)dst[e] * DHID + d], v * w);
}

// pool[batch[r],d] += x[r,d]
__global__ void k_pool(const float* __restrict__ x, const int* __restrict__ batch,
                       float* pool, long total)
{
    long i = (long)blockIdx.x * blockDim.x + threadIdx.x;
    if (i >= total) return;
    long r = i / DHID; int d = (int)(i % DHID);
    atomicAdd(&pool[(long)batch[r] * DHID + d], x[i]);
}

// GRU combine (gi,gh have bih/bhh folded in) -> out = relu((1-z)*n + z*h)
__global__ void k_gru(const float* __restrict__ gi, const float* __restrict__ gh,
                      const float* __restrict__ h, float* out, long rows)
{
    long i = (long)blockIdx.x * blockDim.x + threadIdx.x;
    if (i >= rows * DHID) return;
    long row = i / DHID; int d = (int)(i % DHID);
    const float* gir = gi + row * 288;
    const float* ghr = gh + row * 288;
    float r  = sigmf(gir[d]       + ghr[d]);
    float z  = sigmf(gir[96 + d]  + ghr[96 + d]);
    float nn = tanhf(gir[192 + d] + r * ghr[192 + d]);
    float o  = (1.f - z) * nn + z * h[i];
    out[i] = fmaxf(o, 0.f);
}

// ---------------------------------------------------------------------------
// Host-side orchestration
// ---------------------------------------------------------------------------
static void gemm(hipStream_t s, const float* A, const int* idx, const float* EA,
                 const float* abias, const float* W, const float* bias, float* C,
                 int M, int Nn, int K, int lda, int ldc, int c0,
                 int amode, int pre, int post)
{
    dim3 g((Nn + 63) / 64, (M + 127) / 128);
    k_gemm<<<g, 256, 0, s>>>(A, idx, EA, abias, W, bias, C,
                             M, Nn, K, lda, ldc, c0, amode, pre, post);
}
static inline int cdiv(long a, int b) { return (int)((a + b - 1) / b); }

struct GatP { const float *att_dst, *att_src, *bias, *w; };
struct GruP { const float *bhh, *bih, *whh, *wih; };
struct EncP {
    GatP aconv[3]; GruP agru[3];
    const float *g_attl, *g_attr, *g_bias, *g_w1, *g_w2;
    GruP gru0;
    const float *lin1_b, *lin1_w, *lin2_b, *lin2_w;
    GatP molc; GruP molg;
};
static inline const float* F(void* const* d, int i) { return (const float*)d[i]; }

extern "C" void kernel_launch(void* const* d_in, const int* in_sizes, int n_in,
                              void* d_out, int out_size, void* d_ws, size_t ws_size,
                              hipStream_t stream)
{
    (void)in_sizes; (void)n_in; (void)out_size; (void)ws_size;
    const int N = NNODES, E = NEDGES, G = NGRAPH;

    // ---- graph inputs (dict order: x, edge_index, edge_attr, batch per graph) ----
    const float *X[3], *EATTR[3];
    const int *SRC[3], *DST[3], *BATCH[3];
    for (int g = 0; g < 3; ++g) {
        X[g]     = (const float*)d_in[g * 4 + 0];
        const int* ei = (const int*)d_in[g * 4 + 1];
        SRC[g]   = ei;           // edge_index[0]
        DST[g]   = ei + E;       // edge_index[1]
        EATTR[g] = (const float*)d_in[g * 4 + 2];
        BATCH[g] = (const int*)d_in[g * 4 + 3];
    }

    // ---- params, jax pytree flatten order (sorted dict keys; bind < enc1..3) ----
    int p = 12;
    const float *bb1 = F(d_in, p + 0), *bb2 = F(d_in, p + 1), *bb3 = F(d_in, p + 2);
    const float *bw1 = F(d_in, p + 3), *bw2 = F(d_in, p + 4), *bw3 = F(d_in, p + 5);
    p += 6;
    EncP enc[3];
    for (int i = 0; i < 3; ++i) {
        EncP& e = enc[i];
        for (int l = 0; l < 3; ++l) {   // atom list: conv{att_dst,att_src,bias,w}, gru{bhh,bih,whh,wih}
            e.aconv[l] = { F(d_in, p), F(d_in, p + 1), F(d_in, p + 2), F(d_in, p + 3) }; p += 4;
            e.agru[l]  = { F(d_in, p), F(d_in, p + 1), F(d_in, p + 2), F(d_in, p + 3) }; p += 4;
        }
        e.g_attl = F(d_in, p++); e.g_attr = F(d_in, p++); e.g_bias = F(d_in, p++);
        e.g_w1   = F(d_in, p++); e.g_w2   = F(d_in, p++);
        e.gru0 = { F(d_in, p), F(d_in, p + 1), F(d_in, p + 2), F(d_in, p + 3) }; p += 4;
        e.lin1_b = F(d_in, p++); e.lin1_w = F(d_in, p++);
        e.lin2_b = F(d_in, p++); e.lin2_w = F(d_in, p++);
        e.molc = { F(d_in, p), F(d_in, p + 1), F(d_in, p + 2), F(d_in, p + 3) }; p += 4;
        e.molg = { F(d_in, p), F(d_in, p + 1), F(d_in, p + 2), F(d_in, p + 3) }; p += 4;
    }

    // ---- workspace carve (floats) ----
    float* base = (float*)d_ws;
    float* w_x    = base;                        // N*96
    float* w_h    = w_x  + (long)N * 96;         // N*96
    float* w_xs   = w_h  + (long)N * 96;         // N*96
    float* w_gi   = w_xs + (long)N * 96;         // N*288  (aliases edge buffer)
    float* w_ebuf = w_gi;                        // E*96 <= N*288, disjoint lifetime
    float* w_gh   = w_gi + (long)N * 288;        // N*288
    float* w_alpha  = w_gh + (long)N * 288;      // E
    unsigned* w_smax = (unsigned*)(w_alpha + E); // N
    float* w_ssum = (float*)(w_smax + N);        // N
    float* w_ns1  = w_ssum + N;                  // N
    float* w_ns2  = w_ns1 + N;                   // N
    float* w_pool = w_ns2 + N;                   // G*96
    float* w_mh   = w_pool + (long)G * 96;       // G*96
    float* w_mgi  = w_mh  + (long)G * 96;        // G*288
    float* w_mgh  = w_mgi + (long)G * 288;       // G*288
    float* w_enc  = w_mgh + (long)G * 288;       // G*288 (concat of 3 encoders)
    float* w_b1   = w_enc + (long)G * 288;       // G*1024
    float* w_b2   = w_b1  + (long)G * 1024;      // G*512

    const long NE96 = (long)E * 96, NN96 = (long)N * 96;

    for (int g = 0; g < 3; ++g) {
        const EncP& P = enc[g];
        const int *src = SRC[g], *dst = DST[g];

        // x = leaky(x_in @ lin1_w^T + lin1_b)
        gemm(stream, X[g], nullptr, nullptr, nullptr, P.lin1_w, P.lin1_b, w_x,
             N, 96, 8, 8, 96, 0, 0, 0, 1);

        // ---- GATEConv ----
        // hj = leaky(concat(x[src], edge_attr) @ w1^T)
        gemm(stream, w_x, src, EATTR[g], nullptr, P.g_w1, nullptr, w_ebuf,
             E, 96, 100, 96, 96, 0, 2, 0, 1);
        k_rowdot2<<<cdiv(N, 8), 256, 0, stream>>>(w_x, P.g_attr, nullptr, w_ns1, nullptr, N);
        k_fill_u<<<cdiv(N, 256), 256, 0, stream>>>(w_smax, 0u, N);
        k_gate_score<<<cdiv(E, 8), 256, 0, stream>>>(w_ebuf, P.g_attl, w_ns1, dst,
                                                     w_alpha, w_smax, E);
        // msg = x[src] @ w2^T   (overwrites hj, now dead)
        gemm(stream, w_x, src, nullptr, nullptr, P.g_w2, nullptr, w_ebuf,
             E, 96, 96, 96, 96, 0, 1, 0, 0);
        k_fill_f<<<cdiv(N, 256), 256, 0, stream>>>(w_ssum, 0.f, N);
        k_expsum<<<cdiv(E, 256), 256, 0, stream>>>(w_alpha, w_smax, dst, w_ssum, E);
        k_fill_f<<<cdiv(NN96, 256), 256, 0, stream>>>(w_h, 0.f, NN96);
        k_scatter<<<cdiv(NE96, 256), 256, 0, stream>>>(w_ebuf, nullptr, src, dst,
                                                       w_alpha, w_ssum, w_h, NE96);
        // gi = elu(h + gate_bias) @ wih^T + bih ; gh = x @ whh^T + bhh ; x = relu(GRU)
        gemm(stream, w_h, nullptr, nullptr, P.g_bias, P.gru0.wih, P.gru0.bih, w_gi,
             N, 288, 96, 96, 288, 0, 0, 1, 0);
        gemm(stream, w_x, nullptr, nullptr, nullptr, P.gru0.whh, P.gru0.bhh, w_gh,
             N, 288, 96, 96, 288, 0, 0, 0, 0);
        k_gru<<<cdiv(NN96, 256), 256, 0, stream>>>(w_gi, w_gh, w_x, w_x, N);

        // ---- 3 GATConv atom layers ----
        for (int l = 0; l < 3; ++l) {
            const GatP& cv = P.aconv[l];
            gemm(stream, w_x, nullptr, nullptr, nullptr, cv.w, nullptr, w_xs,
                 N, 96, 96, 96, 96, 0, 0, 0, 0);
            k_rowdot2<<<cdiv(N, 8), 256, 0, stream>>>(w_xs, cv.att_src, cv.att_dst,
                                                      w_ns1, w_ns2, N);
            k_fill_u<<<cdiv(N, 256), 256, 0, stream>>>(w_smax, 0u, N);
            k_gat_score<<<cdiv(E, 256), 256, 0, stream>>>(w_ns1, w_ns2, src, dst,
                                                          w_alpha, w_smax, E);
            k_fill_f<<<cdiv(N, 256), 256, 0, stream>>>(w_ssum, 0.f, N);
            k_expsum<<<cdiv(E, 256), 256, 0, stream>>>(w_alpha, w_smax, dst, w_ssum, E);
            k_fill_f<<<cdiv(NN96, 256), 256, 0, stream>>>(w_h, 0.f, NN96);
            k_scatter<<<cdiv(NE96, 256), 256, 0, stream>>>(nullptr, w_xs, src, dst,
                                                           w_alpha, w_ssum, w_h, NE96);
            gemm(stream, w_h, nullptr, nullptr, cv.bias, P.agru[l].wih, P.agru[l].bih,
                 w_gi, N, 288, 96, 96, 288, 0, 0, 1, 0);
            gemm(stream, w_x, nullptr, nullptr, nullptr, P.agru[l].whh, P.agru[l].bhh,
                 w_gh, N, 288, 96, 96, 288, 0, 0, 0, 0);
            k_gru<<<cdiv(NN96, 256), 256, 0, stream>>>(w_gi, w_gh, w_x, w_x, N);
        }

        // ---- pool + molecule GRU timesteps ----
        const long G96 = (long)G * 96;
        k_fill_f<<<cdiv(G96, 256), 256, 0, stream>>>(w_pool, 0.f, G96);
        k_pool<<<cdiv(NN96, 256), 256, 0, stream>>>(w_x, BATCH[g], w_pool, NN96);
        k_relu<<<cdiv(G96, 256), 256, 0, stream>>>(w_pool, G96);
        for (int t = 0; t < 2; ++t) {
            gemm(stream, w_pool, nullptr, nullptr, nullptr, P.molc.w, P.molc.bias, w_mh,
                 G, 96, 96, 96, 96, 0, 0, 0, 3);                       // elu
            gemm(stream, w_mh, nullptr, nullptr, nullptr, P.molg.wih, P.molg.bih, w_mgi,
                 G, 288, 96, 96, 288, 0, 0, 0, 0);
            gemm(stream, w_pool, nullptr, nullptr, nullptr, P.molg.whh, P.molg.bhh, w_mgh,
                 G, 288, 96, 96, 288, 0, 0, 0, 0);
            k_gru<<<cdiv(G96, 256), 256, 0, stream>>>(w_mgi, w_mgh, w_pool, w_pool, G);
        }
        // lin2 -> concat slice of encoder outputs
        gemm(stream, w_pool, nullptr, nullptr, nullptr, P.lin2_w, P.lin2_b, w_enc,
             G, 96, 96, 96, 288, g * 96, 0, 0, 0);
    }

    // ---- binding MLP ----
    gemm(stream, w_enc, nullptr, nullptr, nullptr, bw1, bb1, w_b1,
         G, 1024, 288, 288, 1024, 0, 0, 0, 2);
    gemm(stream, w_b1, nullptr, nullptr, nullptr, bw2, bb2, w_b2,
         G, 512, 1024, 1024, 512, 0, 0, 0, 2);
    gemm(stream, w_b2, nullptr, nullptr, nullptr, bw3, bb3, (float*)d_out,
         G, 3, 512, 512, 3, 0, 0, 0, 0);
}
